// Attention_31791347925465
// MI455X (gfx1250) — matrix operand order repaired and checked by reading, not compiled, not run on hardware
//
#include <hip/hip_runtime.h>

// ---------------------------------------------------------------------------
// Types for gfx1250 WMMA (wave32)
// ---------------------------------------------------------------------------
typedef __attribute__((ext_vector_type(16))) __bf16        v16bf;
typedef __attribute__((ext_vector_type(8)))  float         v8f;
typedef __attribute__((ext_vector_type(4)))  unsigned int  u32x4;

union AFrag { v16bf bf; u32x4 q[2]; unsigned short us[16]; };
union CFrag { v8f f; float x[8]; };

__device__ __forceinline__ unsigned short f2bf(float f) {
  unsigned int u = __float_as_uint(f);
  u += 0x7FFFu + ((u >> 16) & 1u);          // round-to-nearest-even
  return (unsigned short)(u >> 16);
}
__device__ __forceinline__ float bf2f(unsigned short s) {
  return __uint_as_float(((unsigned int)s) << 16);
}
__device__ __forceinline__ unsigned int pk2(float a, float b) {
  return (unsigned int)f2bf(a) | ((unsigned int)f2bf(b) << 16);
}
__device__ __forceinline__ v8f wmma_bf16(v16bf a, v16bf b, v8f c) {
  // D(16x16,f32) = A(16x32,bf16) x B(32x16,bf16) + C
  return __builtin_amdgcn_wmma_f32_16x16x32_bf16(false, a, false, b, (short)0, c,
                                                 false, false);
}

#define NTOK 4096
#define EMB  768
#define NH   12
#define HD   64
#define RLOG2E 1.44269504088896340736f     // 1/ln(2)

// ---------------------------------------------------------------------------
// Kernel 1: QKV projection  X(4096x768) @ Wqkv(768x2304) + b  -> bf16 QKV
// stored as qkv[which][head][n][d]
// ---------------------------------------------------------------------------
__global__ __launch_bounds__(256) void k_qkv(const float* __restrict__ X,
                                             const float* __restrict__ W,
                                             const float* __restrict__ bias,
                                             unsigned short* __restrict__ qkv) {
  __shared__ unsigned short As[64 * 32];     // [row][k] bf16
  __shared__ unsigned short Bs[64 * 32];     // [col][k] bf16 (B transposed)
  const int tid  = threadIdx.x;
  const int lane = tid & 31, wv = tid >> 5;
  const int hh   = lane >> 4, lc = lane & 15;
  const int rowBase = blockIdx.y * 64;
  const int colBase = blockIdx.x * 64;
  const int rm  = (wv & 3) * 16;             // wave's A-row subtile
  const int cn0 = (wv >> 2) * 32;            // wave's two N subtiles

  CFrag acc0, acc1;
#pragma unroll
  for (int i = 0; i < 8; ++i) { acc0.x[i] = 0.f; acc1.x[i] = 0.f; }

  const int arow = tid >> 2, ac0 = (tid & 3) * 8;
  const int bcol = tid & 63, bk0 = (tid >> 6) * 8;

  for (int k0 = 0; k0 < EMB; k0 += 32) {
    __syncthreads();
    {
      const float* src = X + (size_t)(rowBase + arow) * EMB + k0 + ac0;
      u32x4 pa;
      pa.x = pk2(src[0], src[1]); pa.y = pk2(src[2], src[3]);
      pa.z = pk2(src[4], src[5]); pa.w = pk2(src[6], src[7]);
      *(u32x4*)&As[arow * 32 + ac0] = pa;

      const float* wsrc = W + (size_t)(k0 + bk0) * 2304 + colBase + bcol;
      float wvv[8];
#pragma unroll
      for (int i = 0; i < 8; ++i) wvv[i] = wsrc[(size_t)i * 2304];
      u32x4 pb;
      pb.x = pk2(wvv[0], wvv[1]); pb.y = pk2(wvv[2], wvv[3]);
      pb.z = pk2(wvv[4], wvv[5]); pb.w = pk2(wvv[6], wvv[7]);
      *(u32x4*)&Bs[bcol * 32 + bk0] = pb;
    }
    __syncthreads();

    AFrag a, b0, b1;
    const unsigned short* ap = &As[(rm + lc) * 32 + hh * 16];
    a.q[0] = *(const u32x4*)ap;
    a.q[1] = *(const u32x4*)(ap + 8);
    const unsigned short* bp0 = &Bs[(cn0 + lc) * 32 + hh * 16];
    b0.q[0] = *(const u32x4*)bp0;
    b0.q[1] = *(const u32x4*)(bp0 + 8);
    const unsigned short* bp1 = &Bs[(cn0 + 16 + lc) * 32 + hh * 16];
    b1.q[0] = *(const u32x4*)bp1;
    b1.q[1] = *(const u32x4*)(bp1 + 8);

    acc0.f = wmma_bf16(a.bf, b0.bf, acc0.f);
    acc1.f = wmma_bf16(a.bf, b1.bf, acc1.f);
  }

#pragma unroll
  for (int s2 = 0; s2 < 2; ++s2) {
    CFrag& acc = s2 ? acc1 : acc0;
    const int cn = cn0 + s2 * 16;
    // wave-uniform decomposition: a 16-wide column group never crosses a
    // head (64) boundary, so which/head/dbase are scalar (SALU) values.
    const int cbase = colBase + cn;
    const int which = cbase / EMB;
    const int rem   = cbase - which * EMB;
    const int head  = rem >> 6;
    const int dbase = rem & 63;
    unsigned short* outp =
        qkv + (((size_t)which * NH + head) * NTOK) * HD + dbase + lc;
#pragma unroll
    for (int r = 0; r < 8; ++r) {
      const int n = rowBase + rm + r + hh * 8;
      const float v = acc.x[r] + bias[cbase + lc];
      outp[(size_t)n * HD] = f2bf(v);
    }
  }
}

// ---------------------------------------------------------------------------
// Kernel 2: decomposed rel-pos bias tables (pre-scaled by 1/ln2 so the
// attention kernel can use exp2 directly)
//   Bh[h][n][kh] = dot(q[h,n,:], rel_h[qh-kh+63,:]) / ln2
//   Bw[h][n][kw] = dot(q[h,n,:], rel_w[qw-kw+63,:]) / ln2
// ---------------------------------------------------------------------------
__global__ __launch_bounds__(128) void k_bias(const unsigned short* __restrict__ qkv,
                                              const float* __restrict__ rel_h,
                                              const float* __restrict__ rel_w,
                                              float* __restrict__ Bh,
                                              float* __restrict__ Bw) {
  __shared__ float qrow[HD];
  const int bi = blockIdx.x;
  const int h  = bi >> 12;                   // / 4096
  const int n  = bi & 4095;
  const int qh = n >> 6, qw = n & 63;
  const int t  = threadIdx.x;

  const unsigned short* qp = qkv + ((size_t)h * NTOK + n) * HD;  // which==0 (Q)
  if (t < HD) qrow[t] = bf2f(qp[t]);
  __syncthreads();

  const int kk = t & 63;
  const float* rp = (t < 64) ? (rel_h + (size_t)(qh - kk + 63) * HD)
                             : (rel_w + (size_t)(qw - kk + 63) * HD);
  float s = 0.f;
#pragma unroll 8
  for (int d = 0; d < HD; ++d) s += qrow[d] * rp[d];
  float* dst = (t < 64) ? Bh : Bw;
  dst[((size_t)h * NTOK + n) * HD + kk] = s * RLOG2E;
}

// ---------------------------------------------------------------------------
// Kernel 3: flash attention with folded bias (log2-domain softmax).
// Block = 4 waves x 16 queries = 64 queries; key blocks of 64 == one image
// row, so bias_h is one scalar per (row, block) and bias_w is key-block
// invariant.
// ---------------------------------------------------------------------------
__global__ __launch_bounds__(128) void k_attn(const unsigned short* __restrict__ qkv,
                                              const float* __restrict__ Bh,
                                              const float* __restrict__ Bw,
                                              unsigned short* __restrict__ att) {
  __shared__ unsigned short Ks[64 * 64];     // [key][d]
  __shared__ unsigned short Vt[64 * 64];     // [d][key]  (transposed V)
  __shared__ unsigned short Ps[4 * 16 * 64]; // per-wave P tile [row][key]

  const int tid  = threadIdx.x;
  const int lane = tid & 31, wv = tid >> 5;
  const int hh   = lane >> 4, lc = lane & 15;
  const int h    = blockIdx.y;
  const int qblk = blockIdx.x * 64;

  const unsigned short* Qg = qkv + ((size_t)(0 * NH + h) * NTOK) * HD;
  const unsigned short* Kg = qkv + ((size_t)(1 * NH + h) * NTOK) * HD;
  const unsigned short* Vg = qkv + ((size_t)(2 * NH + h) * NTOK) * HD;

  // Q fragments: kept in registers for whole kernel (2 x 16x32 bf16)
  AFrag qa[2];
  {
    const int qr = qblk + wv * 16 + lc;
#pragma unroll
    for (int t = 0; t < 2; ++t) {
      const unsigned short* p = Qg + (size_t)qr * HD + t * 32 + hh * 16;
      qa[t].q[0] = *(const u32x4*)p;
      qa[t].q[1] = *(const u32x4*)(p + 8);
    }
  }
  // Bw per output element: invariant across key blocks (kw == in-block col)
  float bwreg[4][8];
#pragma unroll
  for (int s2 = 0; s2 < 4; ++s2)
#pragma unroll
    for (int r = 0; r < 8; ++r) {
      const int qr = qblk + wv * 16 + r + hh * 8;
      bwreg[s2][r] = Bw[((size_t)h * NTOK + qr) * HD + s2 * 16 + lc];
    }

  float m[8], l[8];
  CFrag O[4];
#pragma unroll
  for (int r = 0; r < 8; ++r) { m[r] = -1e30f; l[r] = 0.f; }
#pragma unroll
  for (int s2 = 0; s2 < 4; ++s2)
#pragma unroll
    for (int r = 0; r < 8; ++r) O[s2].x[r] = 0.f;

  const int   skey  = tid >> 1, sdh = (tid & 1) * 32;
  const float scale = 0.125f * RLOG2E;        // 64^-0.5 / ln2 (log2 domain)

  for (int j = 0; j < 64; ++j) {              // key blocks (== kh)
    __syncthreads();
    {
      const unsigned short* ks = Kg + ((size_t)(j * 64 + skey)) * HD + sdh;
      const unsigned short* vs = Vg + ((size_t)(j * 64 + skey)) * HD + sdh;
#pragma unroll
      for (int i = 0; i < 4; ++i)
        *(u32x4*)&Ks[skey * 64 + sdh + i * 8] = *(const u32x4*)(ks + i * 8);
      // stage V transposed so P@V fragments read contiguous (b128) runs
      union { u32x4 q[4]; unsigned short us[32]; } vbuf;
#pragma unroll
      for (int i = 0; i < 4; ++i) vbuf.q[i] = *(const u32x4*)(vs + i * 8);
#pragma unroll
      for (int i = 0; i < 32; ++i)
        Vt[(sdh + i) * 64 + skey] = vbuf.us[i];
    }
    __syncthreads();

    // S = Q @ K^T : 16 x 64 per wave (4 subtiles x 2 K-halves)
    CFrag sfr[4];
#pragma unroll
    for (int s2 = 0; s2 < 4; ++s2) {
#pragma unroll
      for (int i = 0; i < 8; ++i) sfr[s2].x[i] = 0.f;
#pragma unroll
      for (int t = 0; t < 2; ++t) {
        AFrag kb;
        const unsigned short* kp = &Ks[(s2 * 16 + lc) * 64 + t * 32 + hh * 16];
        kb.q[0] = *(const u32x4*)kp;
        kb.q[1] = *(const u32x4*)(kp + 8);
        sfr[s2].f = wmma_bf16(qa[t].bf, kb.bf, sfr[s2].f);
      }
    }

    // logits (log2 domain) = scale*S + Bh(row, j) + Bw(row, col)
    float bh_r[8];
#pragma unroll
    for (int r = 0; r < 8; ++r) {
      const int qr = qblk + wv * 16 + r + hh * 8;
      bh_r[r] = Bh[((size_t)h * NTOK + qr) * HD + j];
    }
#pragma unroll
    for (int s2 = 0; s2 < 4; ++s2)
#pragma unroll
      for (int r = 0; r < 8; ++r)
        sfr[s2].x[r] = sfr[s2].x[r] * scale + bh_r[r] + bwreg[s2][r];

    // online softmax: rows live in half-waves; reduce across 16 lanes
    float alpha[8];
#pragma unroll
    for (int r = 0; r < 8; ++r) {
      float v = fmaxf(fmaxf(sfr[0].x[r], sfr[1].x[r]),
                      fmaxf(sfr[2].x[r], sfr[3].x[r]));
      v = fmaxf(v, __shfl_xor(v, 1));
      v = fmaxf(v, __shfl_xor(v, 2));
      v = fmaxf(v, __shfl_xor(v, 4));
      v = fmaxf(v, __shfl_xor(v, 8));
      const float mn = fmaxf(m[r], v);
      alpha[r] = exp2f(m[r] - mn);
      m[r] = mn;
    }
#pragma unroll
    for (int r = 0; r < 8; ++r) {
      float rs = 0.f;
#pragma unroll
      for (int s2 = 0; s2 < 4; ++s2) {
        const float p = exp2f(sfr[s2].x[r] - m[r]);
        sfr[s2].x[r] = p;
        rs += p;
      }
      rs += __shfl_xor(rs, 1);
      rs += __shfl_xor(rs, 2);
      rs += __shfl_xor(rs, 4);
      rs += __shfl_xor(rs, 8);
      l[r] = l[r] * alpha[r] + rs;
#pragma unroll
      for (int s2 = 0; s2 < 4; ++s2) O[s2].x[r] *= alpha[r];
    }

    // P -> bf16 via per-wave LDS tile (C-layout -> A-layout shuffle)
#pragma unroll
    for (int s2 = 0; s2 < 4; ++s2)
#pragma unroll
      for (int r = 0; r < 8; ++r)
        Ps[wv * 1024 + (r + hh * 8) * 64 + s2 * 16 + lc] = f2bf(sfr[s2].x[r]);
    asm volatile("s_wait_dscnt 0" ::: "memory");

    // O += P @ V
    AFrag pa[2];
#pragma unroll
    for (int t = 0; t < 2; ++t) {
      const unsigned short* pp = &Ps[wv * 1024 + lc * 64 + t * 32 + hh * 16];
      pa[t].q[0] = *(const u32x4*)pp;
      pa[t].q[1] = *(const u32x4*)(pp + 8);
    }
#pragma unroll
    for (int s2 = 0; s2 < 4; ++s2) {
#pragma unroll
      for (int t = 0; t < 2; ++t) {
        AFrag vb;
        const unsigned short* vp = &Vt[(s2 * 16 + lc) * 64 + t * 32 + hh * 16];
        vb.q[0] = *(const u32x4*)vp;
        vb.q[1] = *(const u32x4*)(vp + 8);
        O[s2].f = wmma_bf16(pa[t].bf, vb.bf, O[s2].f);
      }
    }
  }

  // epilogue: normalize and store bf16 att[n][h*64+d]
#pragma unroll
  for (int s2 = 0; s2 < 4; ++s2)
#pragma unroll
    for (int r = 0; r < 8; ++r) {
      const int n = qblk + wv * 16 + r + hh * 8;
      const int c = h * HD + s2 * 16 + lc;
      att[(size_t)n * EMB + c] = f2bf(O[s2].x[r] / l[r]);
    }
}

// ---------------------------------------------------------------------------
// Kernel 4: output projection  att(4096x768,bf16) @ Wproj(768x768) + b -> f32
// ---------------------------------------------------------------------------
__global__ __launch_bounds__(256) void k_proj(const unsigned short* __restrict__ A,
                                              const float* __restrict__ W,
                                              const float* __restrict__ bias,
                                              float* __restrict__ out) {
  __shared__ unsigned short As[64 * 32];
  __shared__ unsigned short Bs[64 * 32];
  const int tid  = threadIdx.x;
  const int lane = tid & 31, wv = tid >> 5;
  const int hh   = lane >> 4, lc = lane & 15;
  const int rowBase = blockIdx.y * 64;
  const int colBase = blockIdx.x * 64;
  const int rm  = (wv & 3) * 16;
  const int cn0 = (wv >> 2) * 32;

  CFrag acc0, acc1;
#pragma unroll
  for (int i = 0; i < 8; ++i) { acc0.x[i] = 0.f; acc1.x[i] = 0.f; }

  const int arow = tid >> 2, ac0 = (tid & 3) * 8;
  const int bcol = tid & 63, bk0 = (tid >> 6) * 8;

  for (int k0 = 0; k0 < EMB; k0 += 32) {
    __syncthreads();
    {
      *(u32x4*)&As[arow * 32 + ac0] =
          *(const u32x4*)(A + (size_t)(rowBase + arow) * EMB + k0 + ac0);
      const float* wsrc = W + (size_t)(k0 + bk0) * EMB + colBase + bcol;
      float wvv[8];
#pragma unroll
      for (int i = 0; i < 8; ++i) wvv[i] = wsrc[(size_t)i * EMB];
      u32x4 pb;
      pb.x = pk2(wvv[0], wvv[1]); pb.y = pk2(wvv[2], wvv[3]);
      pb.z = pk2(wvv[4], wvv[5]); pb.w = pk2(wvv[6], wvv[7]);
      *(u32x4*)&Bs[bcol * 32 + bk0] = pb;
    }
    __syncthreads();

    AFrag a, b0, b1;
    const unsigned short* ap = &As[(rm + lc) * 32 + hh * 16];
    a.q[0] = *(const u32x4*)ap;
    a.q[1] = *(const u32x4*)(ap + 8);
    const unsigned short* bp0 = &Bs[(cn0 + lc) * 32 + hh * 16];
    b0.q[0] = *(const u32x4*)bp0;
    b0.q[1] = *(const u32x4*)(bp0 + 8);
    const unsigned short* bp1 = &Bs[(cn0 + 16 + lc) * 32 + hh * 16];
    b1.q[0] = *(const u32x4*)bp1;
    b1.q[1] = *(const u32x4*)(bp1 + 8);

    acc0.f = wmma_bf16(a.bf, b0.bf, acc0.f);
    acc1.f = wmma_bf16(a.bf, b1.bf, acc1.f);
  }

#pragma unroll
  for (int s2 = 0; s2 < 2; ++s2) {
    CFrag& acc = s2 ? acc1 : acc0;
    const int cn = cn0 + s2 * 16;
#pragma unroll
    for (int r = 0; r < 8; ++r) {
      const int n = rowBase + rm + r + hh * 8;
      const int c = colBase + cn + lc;
      out[(size_t)n * EMB + c] = acc.x[r] + bias[c];
    }
  }
}

// ---------------------------------------------------------------------------
// Host-side launch
// ---------------------------------------------------------------------------
extern "C" void kernel_launch(void* const* d_in, const int* in_sizes, int n_in,
                              void* d_out, int out_size, void* d_ws, size_t ws_size,
                              hipStream_t stream) {
  const float* x      = (const float*)d_in[0];
  const float* w_qkv  = (const float*)d_in[1];
  const float* b_qkv  = (const float*)d_in[2];
  const float* w_proj = (const float*)d_in[3];
  const float* b_proj = (const float*)d_in[4];
  const float* rel_h  = (const float*)d_in[5];
  const float* rel_w  = (const float*)d_in[6];
  float* out = (float*)d_out;

  // workspace layout
  unsigned short* qkv = (unsigned short*)d_ws;               // 3*12*4096*64 bf16
  const size_t qkv_elems = (size_t)3 * NH * NTOK * HD;
  float* Bh = (float*)((char*)d_ws + qkv_elems * 2);          // 12*4096*64 f32
  float* Bw = Bh + (size_t)NH * NTOK * HD;                    // 12*4096*64 f32
  unsigned short* att = (unsigned short*)(Bw + (size_t)NH * NTOK * HD); // 4096*768 bf16

  k_qkv <<<dim3(2304 / 64, NTOK / 64), 256, 0, stream>>>(x, w_qkv, b_qkv, qkv);
  k_bias<<<dim3(NH * NTOK),            128, 0, stream>>>(qkv, rel_h, rel_w, Bh, Bw);
  k_attn<<<dim3(NTOK / 64, NH),        128, 0, stream>>>(qkv, Bh, Bw, att);
  k_proj<<<dim3(EMB / 64, NTOK / 64),  256, 0, stream>>>(att, w_proj, b_proj, out);
}
